// RGNNLoss_55602646614219
// MI455X (gfx1250) — matrix operand order repaired and checked by reading, not compiled, not run on hardware
//
#include <hip/hip_runtime.h>
#include <math.h>

// ---------------------------------------------------------------------------
// RGNN pointer-network loss for MI455X (gfx1250, wave32, WMMA).
//   B=512 rows, S=256 steps/slots, H=128, D=2.
// Per-step GEMMs (h@lstm_Wh, h@att_W2) use v_wmma_f32_16x16x32_bf16; the
// loop-invariant B fragments are register-resident (compiler hoists them).
// D=2 lets us fold W_embed into lstm_Wi / att_W1 (2-wide precomputed mats),
// so W1e (64 MB) is never materialized; it is regenerated with 2 FMAs inside
// the tanh-logits loop (v_tanh_f32 transcendental path).
//
// Occupancy: 4 rows/block x 128 blocks (tanh-logits phase dominates; batch
// rows are independent). The WMMA M=16 tile carries 12 zero rows (A rows
// 4..15 = 0 -> D = C = 0), trading cheap matrix ops for 4x parallelism on
// the expensive transcendental phase. LDS ~62 KB/block.
// ---------------------------------------------------------------------------

typedef __attribute__((ext_vector_type(16))) __bf16 v16bf;
typedef __attribute__((ext_vector_type(8)))  float  v8f;

#define H_    128
#define S_    256
#define Br    4       // live batch rows per block
#define NBLK  128     // 512 / 4
#define FOURH 512

// workspace byte offsets (total ~168 KB)
#define WC_OFF    0                  // 1024 f32 : W_embed @ lstm_Wi   (2 x 512)
#define BIASC_OFF 4096               //  512 f32 : lstm_b + b_embed@Wi
#define W1C_OFF   6144               //  256 f32 : W_embed @ att_W1    (2 x 128)
#define B1C_OFF   7168               //  128 f32 : b_embed @ att_W1
#define WHB_OFF   8192               // 65536 u16: lstm_Wh bf16, WMMA-B fragment order
#define W2B_OFF   (8192 + 131072)    // 16384 u16: att_W2  bf16, WMMA-B fragment order

__device__ __forceinline__ unsigned short f2bf(float f) {
  unsigned int u = __float_as_uint(f);
  u += 0x7FFFu + ((u >> 16) & 1u);   // round-to-nearest-even
  return (unsigned short)(u >> 16);
}

__device__ __forceinline__ float tanh_fast(float x) {
#if __has_builtin(__builtin_amdgcn_tanhf)
  return __builtin_amdgcn_tanhf(x);
#else
  float r;
  asm volatile("v_tanh_f32 %0, %1\n\t"
               "s_delay_alu instid0(TRANS32_DEP_1)"
               : "=v"(r) : "v"(x));
  return r;
#endif
}
__device__ __forceinline__ float sig_fast(float x) {
  return 0.5f * tanh_fast(0.5f * x) + 0.5f;
}

// ---------------------------------------------------------------------------
// Prep: zero d_out, fold W_embed/b_embed into lstm_Wi & att_W1, and pre-swizzle
// lstm_Wh / att_W2 into WMMA-B bf16 fragment order:
//   frag[ntile][kc][lane][j]  <=>  B[k = kc*32 + (lane/16)*16 + j][n = ntile*16 + lane%16]
// so the scan kernel loads one contiguous 32B chunk per lane (coalesced).
// ---------------------------------------------------------------------------
__global__ __launch_bounds__(512) void rgnn_prep(
    const float* __restrict__ W_embed, const float* __restrict__ b_embed,
    const float* __restrict__ lstm_Wi, const float* __restrict__ lstm_Wh,
    const float* __restrict__ lstm_b,  const float* __restrict__ att_W1,
    const float* __restrict__ att_W2,  float* __restrict__ dout,
    unsigned char* __restrict__ ws)
{
  const int tid = threadIdx.x;
  float* Wc    = (float*)(ws + WC_OFF);
  float* biasc = (float*)(ws + BIASC_OFF);
  float* W1c   = (float*)(ws + W1C_OFF);
  float* b1c   = (float*)(ws + B1C_OFF);
  unsigned short* WhB = (unsigned short*)(ws + WHB_OFF);
  unsigned short* W2B = (unsigned short*)(ws + W2B_OFF);

  if (tid == 0) dout[0] = 0.0f;   // re-zeroed every launch (graph-replay safe)

  for (int e = tid; e < 1024; e += 512) {
    int d = e >> 9, j = e & 511;
    float s = 0.f;
    for (int h = 0; h < H_; ++h) s = fmaf(W_embed[d*H_ + h], lstm_Wi[h*FOURH + j], s);
    Wc[e] = s;
  }
  for (int j = tid; j < FOURH; j += 512) {
    float s = lstm_b[j];
    for (int h = 0; h < H_; ++h) s = fmaf(b_embed[h], lstm_Wi[h*FOURH + j], s);
    biasc[j] = s;
  }
  for (int e = tid; e < 256; e += 512) {
    int d = e >> 7, j = e & 127;
    float s = 0.f;
    for (int h = 0; h < H_; ++h) s = fmaf(W_embed[d*H_ + h], att_W1[h*H_ + j], s);
    W1c[e] = s;
  }
  for (int j = tid; j < H_; j += 512) {
    float s = 0.f;
    for (int h = 0; h < H_; ++h) s = fmaf(b_embed[h], att_W1[h*H_ + j], s);
    b1c[j] = s;
  }
  // lstm_Wh (128 x 512) -> 32 ntiles x 4 kchunks
  for (int e = tid; e < 32*4*32*16; e += 512) {
    int j = e & 15, lane = (e >> 4) & 31, kc = (e >> 9) & 3, nt = e >> 11;
    int n = nt*16 + (lane & 15);
    int k = kc*32 + ((lane >> 4) << 4) + j;
    WhB[e] = f2bf(lstm_Wh[k*FOURH + n]);
  }
  // att_W2 (128 x 128) -> 8 ntiles x 4 kchunks
  for (int e = tid; e < 8*4*32*16; e += 512) {
    int j = e & 15, lane = (e >> 4) & 31, kc = (e >> 9) & 3, nt = e >> 11;
    int n = nt*16 + (lane & 15);
    int k = kc*32 + ((lane >> 4) << 4) + j;
    W2B[e] = f2bf(att_W2[k*H_ + n]);
  }
}

// ---------------------------------------------------------------------------
// Scan: one block = 4 batch rows, 512 threads = 16 wave32's, 256 steps.
// ---------------------------------------------------------------------------
__global__ __launch_bounds__(512, 1) void rgnn_scan(
    const float* __restrict__ outputs, const float* __restrict__ att_v,
    float* __restrict__ dout, const unsigned char* __restrict__ ws)
{
  __shared__ float h_ld[Br][H_];
  __shared__ float c_ld[Br][H_];
  __shared__ unsigned short hbf[16][H_];     // bf16 A-tile; rows Br..15 stay 0
  __shared__ float gates_ld[16][FOURH];      // rows Br..15 stay 0
  __shared__ float q_ld[16][H_];             // h@att_W2 + b1c (rows Br..15 unused)
  __shared__ float mask_ld[Br][S_];
  __shared__ float Wc_ld[1024];
  __shared__ float biasc_ld[FOURH];
  __shared__ float W1c0[H_], W1c1[H_], b1c_ld[H_], v_ld[H_];
  __shared__ float uav_ld[254][2];
  __shared__ float src_ld[Br][2], dst_ld[Br][2], x_ld[Br][2];
  __shared__ float candV[Br][4];
  __shared__ int   candI[Br][4];
  __shared__ float md_ld[Br];

  const int tid  = threadIdx.x;
  const int lane = tid & 31;
  const int w    = tid >> 5;          // wave id 0..15
  const int b0   = blockIdx.x * Br;

  const float* Wc_g    = (const float*)(ws + WC_OFF);
  const float* biasc_g = (const float*)(ws + BIASC_OFF);
  const float* W1c_g   = (const float*)(ws + W1C_OFF);
  const float* b1c_g   = (const float*)(ws + B1C_OFF);
  const unsigned short* WhB = (const unsigned short*)(ws + WHB_OFF);
  const unsigned short* W2B = (const unsigned short*)(ws + W2B_OFF);

  const float NEG = -__builtin_inff();

  // ---- init ----
  for (int e = tid; e < 1024; e += 512) Wc_ld[e] = Wc_g[e];
  for (int e = tid; e < FOURH; e += 512) biasc_ld[e] = biasc_g[e];
  if (tid < H_) {
    W1c0[tid]   = W1c_g[tid];
    W1c1[tid]   = W1c_g[H_ + tid];
    b1c_ld[tid] = b1c_g[tid];
    v_ld[tid]   = att_v[tid];
  }
  for (int e = tid; e < 508; e += 512) ((float*)uav_ld)[e] = outputs[2048 + e];
  if (tid < Br*2) {
    int b = tid >> 1, d = tid & 1;
    float sv = outputs[(b0 + b)*2 + d];
    src_ld[b][d] = sv;
    x_ld[b][d]   = sv;
    dst_ld[b][d] = outputs[(512 + b0 + b)*2 + d];
  }
  for (int e = tid; e < Br*H_; e += 512)
    if (e < Br*H_) { ((float*)h_ld)[e] = 0.f; ((float*)c_ld)[e] = 0.f; }
  for (int e = tid; e < 16*H_; e += 512) ((unsigned short*)hbf)[e] = 0;
  for (int e = tid; e < 16*FOURH; e += 512) ((float*)gates_ld)[e] = 0.f;
  for (int e = tid; e < Br*S_; e += 512) {
    int s = e & (S_ - 1);
    ((float*)mask_ld)[e] = (s == 0) ? NEG : 0.f;
  }
  __syncthreads();

  // per-row serial state lives in thread `tid` (tid < Br), same thread each step
  float md  = 0.f;
  float y0x = 0.f, y0y = 0.f;
  if (tid < Br) { y0x = src_ld[tid][0]; y0y = src_ld[tid][1]; }

  const int n     = lane & 15;
  const int mb    = (lane >> 4) << 3;   // C/D row offset (0 or 8)
  const int khalf = (lane >> 4) << 3;   // A K offset    (0 or 8)

  for (int k = 0; k < S_; ++k) {
    // phase A: reset last-slot mask; gates(rows 0..Br-1) = x@Wc + biasc
    if (tid < Br) mask_ld[tid][S_ - 1] = (k == 0) ? NEG : 0.f;
    for (int e = tid; e < Br*FOURH; e += 512) {
      int b = e >> 9, j = e & 511;
      gates_ld[b][j] = fmaf(x_ld[b][0], Wc_ld[j],
                       fmaf(x_ld[b][1], Wc_ld[512 + j], biasc_ld[j]));
    }
    __syncthreads();

    // phase B: gates += h @ lstm_Wh  (WMMA bf16, 2 N-tiles per wave)
    for (int t = 0; t < 2; ++t) {
      const int nt = w*2 + t;
      v8f acc;
      #pragma unroll
      for (int r = 0; r < 8; ++r) acc[r] = gates_ld[mb + r][nt*16 + n];
      #pragma unroll
      for (int kc = 0; kc < 4; ++kc) {
        union { v16bf v; unsigned int u[8]; } af;
        const unsigned int* hrow = (const unsigned int*)(&hbf[lane & 15][0]);
        #pragma unroll
        for (int p = 0; p < 8; ++p) {
          int k0 = kc*32 + 2*khalf + 2*p;
          af.u[p] = hrow[k0 >> 1];
        }
        union { v16bf v; unsigned int u[8]; } bfg;
        const unsigned int* bp = (const unsigned int*)(WhB + ((nt*4 + kc)*32 + lane)*16);
        #pragma unroll
        for (int q = 0; q < 8; ++q) bfg.u[q] = bp[q];
        acc = __builtin_amdgcn_wmma_f32_16x16x32_bf16(
                  false, af.v, false, bfg.v, (short)0, acc, false, false);
      }
      #pragma unroll
      for (int r = 0; r < 8; ++r) gates_ld[mb + r][nt*16 + n] = acc[r];
    }
    __syncthreads();

    // phase C: LSTM pointwise on live rows (i,f,g,o split along 4H)
    for (int e = tid; e < Br*H_; e += 512) {
      int b = e >> 7, hh = e & 127;
      float gi = gates_ld[b][hh];
      float gf = gates_ld[b][H_   + hh];
      float gg = gates_ld[b][2*H_ + hh];
      float go = gates_ld[b][3*H_ + hh];
      float cn = sig_fast(gf) * c_ld[b][hh] + sig_fast(gi) * tanh_fast(gg);
      float hn = sig_fast(go) * tanh_fast(cn);
      c_ld[b][hh] = cn;
      h_ld[b][hh] = hn;
      hbf[b][hh]  = f2bf(hn);
    }
    __syncthreads();

    // phase D: q = h @ att_W2 (+ b1c fold), waves 0..7, one N-tile each
    if (w < 8) {
      const int nt = w;
      v8f acc;
      #pragma unroll
      for (int r = 0; r < 8; ++r) acc[r] = 0.f;
      #pragma unroll
      for (int kc = 0; kc < 4; ++kc) {
        union { v16bf v; unsigned int u[8]; } af;
        const unsigned int* hrow = (const unsigned int*)(&hbf[lane & 15][0]);
        #pragma unroll
        for (int p = 0; p < 8; ++p) {
          int k0 = kc*32 + 2*khalf + 2*p;
          af.u[p] = hrow[k0 >> 1];
        }
        union { v16bf v; unsigned int u[8]; } bfg;
        const unsigned int* bp = (const unsigned int*)(W2B + ((nt*4 + kc)*32 + lane)*16);
        #pragma unroll
        for (int q = 0; q < 8; ++q) bfg.u[q] = bp[q];
        acc = __builtin_amdgcn_wmma_f32_16x16x32_bf16(
                  false, af.v, false, bfg.v, (short)0, acc, false, false);
      }
      #pragma unroll
      for (int r = 0; r < 8; ++r)
        q_ld[mb + r][nt*16 + n] = acc[r] + b1c_ld[nt*16 + n];
    }
    __syncthreads();

    // phase E: logits[b][s] = sum_h tanh(W1e + q) * v + mask; argmax (first max)
    // 4 waves per row; wave `w` covers row w&3, s in [ (w>>2)*64, +64 )
    {
      const int b   = w & 3;
      const int qtr = w >> 2;
      const int s0  = qtr * 64;
      float bestV = NEG; int bestI = s0 + lane;
      for (int s = s0 + lane; s < s0 + 64; s += 32) {
        float g0, g1;
        if (s == 0)          { g0 = src_ld[b][0]; g1 = src_ld[b][1]; }
        else if (s == S_ - 1){ g0 = dst_ld[b][0]; g1 = dst_ld[b][1]; }
        else                 { g0 = uav_ld[s-1][0]; g1 = uav_ld[s-1][1]; }
        float accv = 0.f;
        #pragma unroll 4
        for (int hh = 0; hh < H_; ++hh) {
          float t = tanh_fast(fmaf(g0, W1c0[hh], fmaf(g1, W1c1[hh], q_ld[b][hh])));
          accv = fmaf(t, v_ld[hh], accv);
        }
        float logit = accv + mask_ld[b][s];
        if (logit > bestV) { bestV = logit; bestI = s; }   // first max per lane
      }
      #pragma unroll
      for (int off = 16; off > 0; off >>= 1) {
        float ov = __shfl_down(bestV, off, 32);
        int   oi = __shfl_down(bestI, off, 32);
        if (ov > bestV || (ov == bestV && oi < bestI)) { bestV = ov; bestI = oi; }
      }
      if (lane == 0) { candV[b][qtr] = bestV; candI[b][qtr] = bestI; }
    }
    __syncthreads();

    // per-row combine + serial update (thread tid == row), quarters ascending
    if (tid < Br) {
      const int b = tid;
      float bestV = candV[b][0]; int bestI = candI[b][0];
      #pragma unroll
      for (int qq = 1; qq < 4; ++qq) {
        float ov = candV[b][qq]; int oi = candI[b][qq];
        if (ov > bestV) { bestV = ov; bestI = oi; }   // ascending s -> strict
      }
      float y1x, y1y;
      if (bestI == 0)          { y1x = src_ld[b][0]; y1y = src_ld[b][1]; }
      else if (bestI == S_ - 1){ y1x = dst_ld[b][0]; y1y = dst_ld[b][1]; }
      else                     { y1x = uav_ld[bestI-1][0]; y1y = uav_ld[bestI-1][1]; }
      float dx = y1x - y0x, dy = y1y - y0y;
      md = fmaxf(md, sqrtf(fmaf(dx, dx, fmaf(dy, dy, 1e-12f))));
      mask_ld[b][bestI] += NEG;        // .add(NEG)
      x_ld[b][0] = y1x; x_ld[b][1] = y1y;
      y0x = y1x; y0y = y1y;
    }
    __syncthreads();
  }

  if (tid < Br) md_ld[tid] = md;
  __syncthreads();
  if (tid == 0) {
    float s = 0.f;
    #pragma unroll
    for (int i = 0; i < Br; ++i) s += md_ld[i];
    atomicAdd(dout, s * (1.0f / 512.0f));   // mean over B=512
  }
}

extern "C" void kernel_launch(void* const* d_in, const int* in_sizes, int n_in,
                              void* d_out, int out_size, void* d_ws, size_t ws_size,
                              hipStream_t stream)
{
  const float* outputs = (const float*)d_in[0];
  const float* W_embed = (const float*)d_in[1];
  const float* b_embed = (const float*)d_in[2];
  const float* lstm_Wi = (const float*)d_in[3];
  const float* lstm_Wh = (const float*)d_in[4];
  const float* lstm_b  = (const float*)d_in[5];
  const float* att_W1  = (const float*)d_in[6];
  const float* att_W2  = (const float*)d_in[7];
  const float* att_v   = (const float*)d_in[8];
  (void)in_sizes; (void)n_in; (void)out_size; (void)ws_size;

  float* dout = (float*)d_out;
  unsigned char* ws = (unsigned char*)d_ws;   // needs ~172 KB

  rgnn_prep<<<1, 512, 0, stream>>>(W_embed, b_embed, lstm_Wi, lstm_Wh,
                                   lstm_b, att_W1, att_W2, dout, ws);
  rgnn_scan<<<NBLK, 512, 0, stream>>>(outputs, att_v, dout, ws);
}
